// Model_82995948028470
// MI455X (gfx1250) — matrix operand order repaired and checked by reading, not compiled, not run on hardware
//
#include <hip/hip_runtime.h>
#include <hip/hip_bf16.h>

typedef __attribute__((ext_vector_type(16))) _Float16 v16h;
typedef __attribute__((ext_vector_type(8)))  _Float16 v8h;
typedef __attribute__((ext_vector_type(8)))  float    v8f;

static constexpr int kB = 64;   // batch
static constexpr int kS = 512;  // sequence
static constexpr int kT = 128;  // tags

__device__ __forceinline__ float fast_exp(float x) {   // raw v_exp_f32
    return __builtin_amdgcn_exp2f(x * 1.44269504088896340736f);
}
__device__ __forceinline__ float fast_log(float x) {   // raw v_log_f32
    return __builtin_amdgcn_logf(x) * 0.69314718055994530942f;
}

// ---------------------------------------------------------------------------
// Kernel 1: CRF forward algorithm (the sequential logsumexp scan).
// Grid: 4 blocks (16 batches each), 256 threads = 8 waves (16 next-tags each).
// Per step: logsumexp contraction done as f16 WMMA on exp-shifted operands.
// 2 workgroup barriers per time step; row-max via wave32 shuffles; each lane
// keeps its 8 partition cells in registers (LDS copy is publish-only);
// mask applied as a branchless arithmetic blend so the 8 v_log_f32 pipeline.
// ---------------------------------------------------------------------------
__global__ __launch_bounds__(256)
void crf_forward_kernel(const float* __restrict__ feats,
                        const int*   __restrict__ mask,
                        const float* __restrict__ trans,
                        const float* __restrict__ start_t,
                        const float* __restrict__ stop_t,
                        float* __restrict__ fwd_out)
{
    __shared__ alignas(16) float    part[16][kT];   // partition state (f32)
    __shared__ alignas(16) _Float16 Eld[16][kT];    // exp(partition - pmax) (f16)
    __shared__ alignas(16) float    pmax[16];       // per-batch running max
    __shared__ alignas(16) float    mrow[16];       // mask[b, t] as float

    const int tid   = threadIdx.x;
    const int lane  = tid & 31;
    const int wave  = tid >> 5;          // 0..7 -> N-tile (16 next-tags)
    const int b0    = blockIdx.x * 16;   // batch tile base
    const int khalf = lane >> 4;         // which K-half of the fragment this lane holds
    const int nCol  = wave * 16 + (lane & 15);   // next-tag column this lane serves
    const int row   = tid >> 4;          // 0..15 local batch (phase A mapping)
    const int cb    = (tid & 15) * 8;    // 8-column chunk (phase A mapping)

    // ---- Build persistent B fragments: Bm[i,n] = exp(trans[n,i] - rmax[n]) ----
    // B layout (16-bit 32x16, wave32): lane = N; lanes 0-15 hold K=base..base+15,
    // lanes 16-31 hold K=base+16..base+31; element e <-> K offset e (contiguous).
    float rmaxN = -3.402823e38f;
    for (int c = 0; c < 4; ++c) {
        const float* rp = trans + (size_t)nCol * kT + c * 32 + khalf * 16;
        #pragma unroll
        for (int k = 0; k < 16; ++k) rmaxN = fmaxf(rmaxN, rp[k]);
    }
    rmaxN = fmaxf(rmaxN, __shfl_xor(rmaxN, 16, 32));   // combine the two K-halves

    v16h Bfrag[4];
    for (int c = 0; c < 4; ++c) {
        const float* rp = trans + (size_t)nCol * kT + c * 32 + khalf * 16;
        v16h bf;
        #pragma unroll
        for (int k = 0; k < 16; ++k) bf[k] = (_Float16)fast_exp(rp[k] - rmaxN);
        Bfrag[c] = bf;
    }

    // ---- Initialize partition: feats[:,0,:] + start_transitions ----
    #pragma unroll
    for (int k = 0; k < 8; ++k) {
        const int col = cb + k;
        part[row][col] = feats[((size_t)(b0 + row) * kS + 0) * kT + col] + start_t[col];
    }
    __syncthreads();

    // Each lane owns partition cells (bl = 8*khalf + r, nCol): cache in registers.
    float pold[8];
    #pragma unroll
    for (int r = 0; r < 8; ++r) pold[r] = part[8 * khalf + r][nCol];

    // ---- Sequential scan over time ----
    for (int t = 1; t < kS; ++t) {
        // Phase A: row max via wave shuffles (16-lane groups), then
        // E = exp(part - pmax) as f16 into LDS. No extra barriers.
        float v[8];
        float m = -3.402823e38f;
        #pragma unroll
        for (int k = 0; k < 8; ++k) { v[k] = part[row][cb + k]; m = fmaxf(m, v[k]); }
        #pragma unroll
        for (int off = 8; off >= 1; off >>= 1)
            m = fmaxf(m, __shfl_xor(m, off, 32));      // stays within 16-lane half
        if ((tid & 15) == 0) {
            pmax[row] = m;
            mrow[row] = (float)mask[(size_t)(b0 + row) * kS + t];
        }
        if (t + 1 < kS)   // prefetch next step's feats tile (gfx1250 global_prefetch)
            __builtin_prefetch(&feats[((size_t)(b0 + row) * kS + (t + 1)) * kT + cb], 0, 3);
        #pragma unroll
        for (int k = 0; k < 8; ++k) Eld[row][cb + k] = (_Float16)fast_exp(v[k] - m);
        __syncthreads();

        // Issue phase-C feats loads early so VMEM latency overlaps the WMMAs.
        float ffeat[8];
        #pragma unroll
        for (int r = 0; r < 8; ++r) {
            const int bl = 8 * khalf + r;
            ffeat[r] = feats[((size_t)(b0 + bl) * kS + t) * kT + nCol];
        }
        // Batched reads of this lane's pmax / mask values (2 x b128 each).
        const float4 p0 = *(const float4*)&pmax[8 * khalf];
        const float4 p1 = *(const float4*)&pmax[8 * khalf + 4];
        const float4 q0 = *(const float4*)&mrow[8 * khalf];
        const float4 q1 = *(const float4*)&mrow[8 * khalf + 4];
        const float pmv[8] = {p0.x, p0.y, p0.z, p0.w, p1.x, p1.y, p1.z, p1.w};
        const float mrv[8] = {q0.x, q0.y, q0.z, q0.w, q1.x, q1.y, q1.z, q1.w};

        // Phase B: acc[16x16] = E(16x128) x Bm(128x16) via 4 x wmma_f32_16x16x32_f16.
        // A layout (16-bit 16x32): lane M = lane&15; lanes 0-15 hold K=base..base+15,
        // lanes 16-31 hold K=base+16..base+31 (contiguous, matching B).
        v8f acc = {};
        #pragma unroll
        for (int c = 0; c < 4; ++c) {
            const _Float16* erow = &Eld[lane & 15][0];
            v8h lo = *(const v8h*)(erow + c * 32 + khalf * 16);
            v8h hi = *(const v8h*)(erow + c * 32 + khalf * 16 + 8);
            v16h a = __builtin_shufflevector(lo, hi, 0, 1, 2, 3, 4, 5, 6, 7,
                                                     8, 9, 10, 11, 12, 13, 14, 15);
            acc = __builtin_amdgcn_wmma_f32_16x16x32_f16(
                false, a, false, Bfrag[c], (short)0, acc, false, false);
        }

        // Phase C: new partition = feat + pmax[b] + rmax[n] + log(acc); mask as a
        // branchless blend (pold += m * (cur - pold)) so logs pipeline and no
        // exec-mask churn. C/D layout: VGPR r -> (M = r + 8*khalf, N = lane&15).
        #pragma unroll
        for (int r = 0; r < 8; ++r) {
            const float cur = ffeat[r] + pmv[r] + rmaxN + fast_log(acc[r]);
            pold[r] = fmaf(mrv[r], cur - pold[r], pold[r]);
            part[8 * khalf + r][nCol] = pold[r];
        }
        __syncthreads();
    }

    // ---- Final: fwd[b] = logsumexp_j(partition[b,j] + stop_transitions[j]) ----
    if (tid < 16) {
        float m = -3.402823e38f;
        for (int j = 0; j < kT; ++j) m = fmaxf(m, part[tid][j] + stop_t[j]);
        float s = 0.0f;
        for (int j = 0; j < kT; ++j) s += fast_exp(part[tid][j] + stop_t[j] - m);
        fwd_out[b0 + tid] = m + fast_log(s);
    }
}

// ---------------------------------------------------------------------------
// Kernel 2: gold-path score (gather + reduce). One block per batch.
// ---------------------------------------------------------------------------
__global__ __launch_bounds__(128)
void crf_gold_kernel(const float* __restrict__ feats,
                     const int*   __restrict__ mask,
                     const int*   __restrict__ tags,
                     const float* __restrict__ trans,
                     const float* __restrict__ start_t,
                     const float* __restrict__ stop_t,
                     float* __restrict__ gold_out)
{
    __shared__ float rs[128];
    __shared__ int   rl[128];
    const int b   = blockIdx.x;
    const int tid = threadIdx.x;

    float acc = 0.0f;
    int   len = 0;
    for (int t = tid; t < kS; t += 128) {
        const int   mk = mask[(size_t)b * kS + t];
        const float mf = (float)mk;
        const int   tg = tags[(size_t)b * kS + t];
        acc += feats[((size_t)b * kS + t) * kT + tg] * mf;
        if (t >= 1) {
            const int tp = tags[(size_t)b * kS + t - 1];
            acc += trans[(size_t)tp * kT + tg] * mf;
        }
        len += mk;
    }
    rs[tid] = acc;
    rl[tid] = len;
    __syncthreads();
    for (int s = 64; s > 0; s >>= 1) {
        if (tid < s) { rs[tid] += rs[tid + s]; rl[tid] += rl[tid + s]; }
        __syncthreads();
    }
    if (tid == 0) {
        const int L = rl[0];
        gold_out[b] = rs[0] + start_t[tags[(size_t)b * kS]]
                            + stop_t[tags[(size_t)b * kS + (L - 1)]];
    }
}

// ---------------------------------------------------------------------------
// Kernel 3: scalar loss = sum_b (fwd[b] - gold[b])
// ---------------------------------------------------------------------------
__global__ __launch_bounds__(64)
void crf_final_kernel(const float* __restrict__ fwd,
                      const float* __restrict__ gold,
                      float* __restrict__ out)
{
    __shared__ float rs[64];
    const int tid = threadIdx.x;
    rs[tid] = fwd[tid] - gold[tid];
    __syncthreads();
    for (int s = 32; s > 0; s >>= 1) {
        if (tid < s) rs[tid] += rs[tid + s];
        __syncthreads();
    }
    if (tid == 0) out[0] = rs[0];
}

// ---------------------------------------------------------------------------
extern "C" void kernel_launch(void* const* d_in, const int* in_sizes, int n_in,
                              void* d_out, int out_size, void* d_ws, size_t ws_size,
                              hipStream_t stream) {
    const float* feats   = (const float*)d_in[0];   // (64,512,128) f32
    const int*   mask    = (const int*)  d_in[1];   // (64,512) i32
    const int*   tags    = (const int*)  d_in[2];   // (64,512) i32
    const float* trans   = (const float*)d_in[3];   // (128,128) f32
    const float* start_t = (const float*)d_in[4];   // (128,) f32
    const float* stop_t  = (const float*)d_in[5];   // (128,) f32
    float* out = (float*)d_out;

    float* fwd  = (float*)d_ws;        // 64 floats
    float* gold = (float*)d_ws + 64;   // 64 floats

    crf_forward_kernel<<<kB / 16, 256, 0, stream>>>(feats, mask, trans, start_t, stop_t, fwd);
    crf_gold_kernel<<<kB, 128, 0, stream>>>(feats, mask, tags, trans, start_t, stop_t, gold);
    crf_final_kernel<<<1, 64, 0, stream>>>(fwd, gold, out);
}
